// MultiHeadAreaAttention_54743653155065
// MI455X (gfx1250) — matrix operand order repaired and checked by reading, not compiled, not run on hardware
//
#include <hip/hip_runtime.h>

// ---------------------------------------------------------------------------
// MultiHeadAreaAttention for MI455X (gfx1250, wave32, WMMA f32_16x16x32_f16)
// ---------------------------------------------------------------------------
// B=16 L=512 D_MODEL=512 H=8 DH=DV=64 W=3  -> A = 1533 areas (pad to 1536)
//
// ws layout (in _Float16 elements):
//   qh  [128][512][64]      f16   @ 0
//   kh  [128][512][64]      f16   @ 4194304
//   vh  [128][512][64]      f16   @ 8388608
//   ka  [128][1536][64]     f16   @ 12582912   (area-mean keys, padded A)
//   va  [128][1536][64]     f16   @ 25165824   (area-sum values, padded A)
//   at  [16][512][512]      f16   @ 37748736   (attention out, head-merged)
//   wt  [4][512][512]       f16   @ 41943040   (Wq^T,Wk^T,Wv^T,Wo^T as f16)
// total 42991616 halves = 82 MiB workspace
// ---------------------------------------------------------------------------

typedef __attribute__((ext_vector_type(16))) _Float16 v16h;
typedef __attribute__((ext_vector_type(8)))  float    v8f;

#define HH      8
#define AW      1533
#define APAD    1536
#define NEGINF  (-1e30f)

#define OFF_QH  0ull
#define OFF_KH  4194304ull
#define OFF_VH  8388608ull
#define OFF_KA  12582912ull
#define OFF_VA  25165824ull
#define OFF_AT  37748736ull
#define OFF_WT  41943040ull
#define WT_SZ   262144ull      // halves per transposed 512x512 matrix

// A-matrix (16x32 f16) per-lane K index for element j:
// lanes 0-15 hold K=0..15; lanes 16-31 hold K=16..31 (contiguous, matches B).
__device__ __forceinline__ int kmapA(int j, int g) {
  return j + 16 * g;
}

// end position of area index a (w=1: [0,512), w=2: [512,1023), w=3: [1023,1533))
__device__ __forceinline__ int area_end(int a) {
  return (a < 512) ? a : ((a < 1023) ? (a - 511) : (a - 1021));
}

// ---------------------------------------------------------------------------
// Kernel 0: one-time weight transpose+convert.  WT[mtx][e][c] = (f16)W[c][e].
// Makes every GEMM B-fragment a contiguous 32B load.
// ---------------------------------------------------------------------------
__global__ void __launch_bounds__(256) wtrans_kernel(
    const float* __restrict__ Wq, const float* __restrict__ Wk,
    const float* __restrict__ Wv, const float* __restrict__ Wo,
    _Float16* __restrict__ wt)
{
  const int mtx = blockIdx.y;
  const float* W = (mtx == 0) ? Wq : (mtx == 1) ? Wk : (mtx == 2) ? Wv : Wo;
  const int tid = blockIdx.x * 256 + threadIdx.x;   // 262144 threads per matrix
  const int c = tid >> 9;        // input feature (row of W)
  const int e = tid & 511;       // output feature (col of W)
  wt[(size_t)mtx * WT_SZ + (size_t)e * 512 + c] = (_Float16)W[(size_t)c * 512 + e];
}

// ---------------------------------------------------------------------------
// Kernel 1: fused QKV projection GEMM.  One wave per 16x16 output tile.
// X[8192,512](f32) x W^T(f16) + bias -> head-split f16 [128][512][64]
// ---------------------------------------------------------------------------
__global__ void __launch_bounds__(32) qkv_proj_kernel(
    const float* __restrict__ q, const float* __restrict__ k, const float* __restrict__ v,
    const float* __restrict__ bq, const float* __restrict__ bk, const float* __restrict__ bv,
    _Float16* __restrict__ ws)
{
  const int z = blockIdx.z;
  const float* X; const float* bias; _Float16* Y;
  if (z == 0)      { X = q; bias = bq; Y = ws + OFF_QH; }
  else if (z == 1) { X = k; bias = bk; Y = ws + OFF_KH; }
  else             { X = v; bias = bv; Y = ws + OFF_VH; }
  const _Float16* WT = ws + OFF_WT + (size_t)z * WT_SZ;

  const int lane = threadIdx.x & 31;
  const int m = lane & 15;
  const int g = lane >> 4;
  const int mt = blockIdx.y, nt = blockIdx.x;
  const int row = mt * 16 + m;   // row in [0, B*L)
  const int n   = nt * 16 + m;   // output feature

  const _Float16* wrow = WT + (size_t)n * 512 + 16 * g;  // lane's B column, contiguous K

  v8f c = {};
  for (int kc = 0; kc < 16; ++kc) {
    const int k0 = kc * 32;
    v16h a;
#pragma unroll
    for (int j = 0; j < 16; ++j)
      a[j] = (_Float16)X[(size_t)row * 512 + k0 + kmapA(j, g)];
    const v16h b = *(const v16h*)(wrow + k0);
    c = __builtin_amdgcn_wmma_f32_16x16x32_f16(false, a, false, b, (short)0, c, false, false);
  }

  const float bb = bias[n];
  const int h = n >> 6, d = n & 63;
#pragma unroll
  for (int i = 0; i < 8; ++i) {
    const int r = mt * 16 + i + 8 * g;          // global row
    const int bidx = r >> 9, lpos = r & 511;
    Y[(((size_t)(bidx * HH + h)) * 512 + lpos) * 64 + d] = (_Float16)(c[i] + bb);
  }
}

// ---------------------------------------------------------------------------
// Kernel 2: area pooling via running cumsum. One thread per (bh, d): 8192 total.
// k_area = window-mean of kh, v_area = window-sum of vh, widths 1..3.
// ---------------------------------------------------------------------------
__global__ void __launch_bounds__(256) area_feat_kernel(_Float16* __restrict__ ws)
{
  const int tid = blockIdx.x * 256 + threadIdx.x;
  const int bh = tid >> 6;
  const int d  = tid & 63;
  const _Float16* kh = ws + OFF_KH + ((size_t)bh * 512) * 64 + d;
  const _Float16* vh = ws + OFF_VH + ((size_t)bh * 512) * 64 + d;
  _Float16* ka = ws + OFF_KA + (size_t)bh * APAD * 64 + d;
  _Float16* va = ws + OFF_VA + (size_t)bh * APAD * 64 + d;

  float pk0 = 0.f, pk1 = 0.f, pk2 = 0.f;
  float pv0 = 0.f, pv1 = 0.f, pv2 = 0.f;
  for (int l = 0; l < 512; ++l) {
    const float ck = pk0 + (float)kh[(size_t)l * 64];
    const float cv = pv0 + (float)vh[(size_t)l * 64];
    // w = 1, area index l
    ka[(size_t)l * 64] = (_Float16)(ck - pk0);
    va[(size_t)l * 64] = (_Float16)(cv - pv0);
    // w = 2, area index 512 + (l-1)
    if (l >= 1) {
      ka[(size_t)(512 + l - 1) * 64] = (_Float16)((ck - pk1) * 0.5f);
      va[(size_t)(512 + l - 1) * 64] = (_Float16)(cv - pv1);
    }
    // w = 3, area index 1023 + (l-2)
    if (l >= 2) {
      ka[(size_t)(1023 + l - 2) * 64] = (_Float16)((ck - pk2) * (1.0f / 3.0f));
      va[(size_t)(1023 + l - 2) * 64] = (_Float16)(cv - pv2);
    }
    pk2 = pk1; pk1 = pk0; pk0 = ck;
    pv2 = pv1; pv1 = pv0; pv0 = cv;
  }
}

// ---------------------------------------------------------------------------
// Kernel 3: flash-style area attention. One wave per (bh, 16-query tile).
// Iterates 48 blocks of 32 areas: 4 WMMAs for logits, online softmax with
// __shfl_xor row reductions in the 16-lane half-groups, P staged through LDS
// into A-layout, 4 WMMAs for P x V_area.  Causal block-skip is uniform.
// ---------------------------------------------------------------------------
__global__ void __launch_bounds__(128) area_attn_kernel(_Float16* __restrict__ ws)
{
  __shared__ _Float16 pbuf[4][16][32];   // per-wave P staging (1 KB each)

  const int lane = threadIdx.x & 31;
  const int wid  = threadIdx.x >> 5;
  const int qt   = blockIdx.x * 4 + wid;   // query tile 0..31
  const int bh   = blockIdx.y;             // 0..127
  const int m = lane & 15, g = lane >> 4;

  const _Float16* qh = ws + OFF_QH;
  const _Float16* ka = ws + OFF_KA + (size_t)bh * APAD * 64;
  const _Float16* va = ws + OFF_VA + (size_t)bh * APAD * 64;

  // Q fragments (row = lane&15), K=64 split in two 32-chunks
  v16h qf0, qf1;
  {
    const _Float16* qr = qh + ((size_t)bh * 512 + qt * 16 + m) * 64;
#pragma unroll
    for (int j = 0; j < 16; ++j) qf0[j] = qr[kmapA(j, g)];
#pragma unroll
    for (int j = 0; j < 16; ++j) qf1[j] = qr[32 + kmapA(j, g)];
  }

  v8f acc0 = {}, acc1 = {}, acc2 = {}, acc3 = {};
  float rm[8], rl[8];
#pragma unroll
  for (int i = 0; i < 8; ++i) { rm[i] = -3.0e38f; rl[i] = 0.f; }

  const int qmax = qt * 16 + 15;

  for (int ab = 0; ab < 48; ++ab) {
    const int a0 = ab * 32;
    // minimum area-end in this 32-area block (scalar/uniform): skip if causal-dead
    int minEnd = 1 << 30;
    {
      int lo, hi, e;
      lo = a0 > 0 ? a0 : 0;        hi = a0 + 31 < 511  ? a0 + 31 : 511;
      if (lo <= hi) { e = lo;        minEnd = e < minEnd ? e : minEnd; }
      lo = a0 > 512 ? a0 : 512;    hi = a0 + 31 < 1022 ? a0 + 31 : 1022;
      if (lo <= hi) { e = lo - 511;  minEnd = e < minEnd ? e : minEnd; }
      lo = a0 > 1023 ? a0 : 1023;  hi = a0 + 31 < 1532 ? a0 + 31 : 1532;
      if (lo <= hi) { e = lo - 1021; minEnd = e < minEnd ? e : minEnd; }
    }
    if (minEnd > qmax) continue;

    // ---- logits S = Q x K_area^T for 32 areas (two 16-col tiles) ----
    v8f s0 = {}, s1 = {};
    {
      const _Float16* kr0 = ka + (size_t)(a0 + m) * 64 + 16 * g;
      const _Float16* kr1 = ka + (size_t)(a0 + 16 + m) * 64 + 16 * g;
      const v16h b0 = *(const v16h*)(kr0);
      const v16h b1 = *(const v16h*)(kr0 + 32);
      const v16h b2 = *(const v16h*)(kr1);
      const v16h b3 = *(const v16h*)(kr1 + 32);
      s0 = __builtin_amdgcn_wmma_f32_16x16x32_f16(false, qf0, false, b0, (short)0, s0, false, false);
      s0 = __builtin_amdgcn_wmma_f32_16x16x32_f16(false, qf1, false, b1, (short)0, s0, false, false);
      s1 = __builtin_amdgcn_wmma_f32_16x16x32_f16(false, qf0, false, b2, (short)0, s1, false, false);
      s1 = __builtin_amdgcn_wmma_f32_16x16x32_f16(false, qf1, false, b3, (short)0, s1, false, false);
    }

    // ---- mask + online softmax ----
    float nm[8], sc[8];
#pragma unroll
    for (int i = 0; i < 8; ++i) {
      const int qpos = qt * 16 + i + 8 * g;
      const int aA = a0 + m;
      const int aB = a0 + 16 + m;
      float x0 = s0[i], x1 = s1[i];
      x0 = ((aA < AW) && (area_end(aA) <= qpos)) ? x0 : NEGINF;
      x1 = ((aB < AW) && (area_end(aB) <= qpos)) ? x1 : NEGINF;
      s0[i] = x0; s1[i] = x1;
      float v2 = fmaxf(x0, x1);
#pragma unroll
      for (int off = 1; off < 16; off <<= 1)
        v2 = fmaxf(v2, __shfl_xor(v2, off, 32));
      nm[i] = fmaxf(rm[i], v2);
      sc[i] = __expf(rm[i] - nm[i]);
      rm[i] = nm[i];
    }
#pragma unroll
    for (int i = 0; i < 8; ++i) {
      const float e0 = __expf(s0[i] - nm[i]);
      const float e1 = __expf(s1[i] - nm[i]);
      s0[i] = e0; s1[i] = e1;
      float rs = e0 + e1;
#pragma unroll
      for (int off = 1; off < 16; off <<= 1)
        rs += __shfl_xor(rs, off, 32);
      rl[i] = rl[i] * sc[i] + rs;
      acc0[i] *= sc[i]; acc1[i] *= sc[i]; acc2[i] *= sc[i]; acc3[i] *= sc[i];
    }

    // ---- convert P from C-layout to A-layout via per-wave LDS slice ----
#pragma unroll
    for (int i = 0; i < 8; ++i) {
      pbuf[wid][i + 8 * g][m]      = (_Float16)s0[i];
      pbuf[wid][i + 8 * g][16 + m] = (_Float16)s1[i];
    }
    v16h pf;
#pragma unroll
    for (int j = 0; j < 16; ++j) pf[j] = pbuf[wid][m][kmapA(j, g)];

    // ---- P x V_area (K = 32 areas, four 16-wide dv tiles) ----
    {
      const _Float16* vr = va + (size_t)(a0 + 16 * g) * 64;
      v16h b0, b1, b2, b3;
#pragma unroll
      for (int j = 0; j < 16; ++j) {
        const _Float16* vrow = vr + (size_t)j * 64;
        b0[j] = vrow[m];
        b1[j] = vrow[16 + m];
        b2[j] = vrow[32 + m];
        b3[j] = vrow[48 + m];
      }
      acc0 = __builtin_amdgcn_wmma_f32_16x16x32_f16(false, pf, false, b0, (short)0, acc0, false, false);
      acc1 = __builtin_amdgcn_wmma_f32_16x16x32_f16(false, pf, false, b1, (short)0, acc1, false, false);
      acc2 = __builtin_amdgcn_wmma_f32_16x16x32_f16(false, pf, false, b2, (short)0, acc2, false, false);
      acc3 = __builtin_amdgcn_wmma_f32_16x16x32_f16(false, pf, false, b3, (short)0, acc3, false, false);
    }
  }

  // ---- normalize + write head-merged attention output [B][L][H*DV] f16 ----
  {
    const int bb2 = bh >> 3, hh2 = bh & 7;
    _Float16* at = ws + OFF_AT;
#pragma unroll
    for (int i = 0; i < 8; ++i) {
      const int qpos = qt * 16 + i + 8 * g;
      const float inv = 1.0f / rl[i];
      const size_t base = ((size_t)(bb2 * 512 + qpos)) * 512 + hh2 * 64 + m;
      at[base]      = (_Float16)(acc0[i] * inv);
      at[base + 16] = (_Float16)(acc1[i] * inv);
      at[base + 32] = (_Float16)(acc2[i] * inv);
      at[base + 48] = (_Float16)(acc3[i] * inv);
    }
  }
}

// ---------------------------------------------------------------------------
// Kernel 4: output projection. attn[8192,512] f16 x Wo^T(f16) + bo -> f32 out
// ---------------------------------------------------------------------------
__global__ void __launch_bounds__(32) out_proj_kernel(
    const _Float16* __restrict__ ws_ro, const float* __restrict__ bo,
    float* __restrict__ out)
{
  const _Float16* at = ws_ro + OFF_AT;
  const _Float16* WT = ws_ro + OFF_WT + 3 * WT_SZ;   // Wo^T

  const int lane = threadIdx.x & 31;
  const int m = lane & 15, g = lane >> 4;
  const int mt = blockIdx.y, nt = blockIdx.x;
  const int row = mt * 16 + m;
  const int n   = nt * 16 + m;

  const _Float16* wrow = WT + (size_t)n * 512 + 16 * g;

  v8f c = {};
  for (int kc = 0; kc < 16; ++kc) {
    const int k0 = kc * 32;
    v16h a;
#pragma unroll
    for (int j = 0; j < 16; ++j)
      a[j] = at[(size_t)row * 512 + k0 + kmapA(j, g)];
    const v16h b = *(const v16h*)(wrow + k0);
    c = __builtin_amdgcn_wmma_f32_16x16x32_f16(false, a, false, b, (short)0, c, false, false);
  }

  const float bb = bo[n];
#pragma unroll
  for (int i = 0; i < 8; ++i) {
    const int r = mt * 16 + i + 8 * g;
    out[(size_t)r * 512 + n] = c[i] + bb;
  }
}

// ---------------------------------------------------------------------------
extern "C" void kernel_launch(void* const* d_in, const int* in_sizes, int n_in,
                              void* d_out, int out_size, void* d_ws, size_t ws_size,
                              hipStream_t stream)
{
  (void)in_sizes; (void)n_in; (void)out_size; (void)ws_size;
  const float* q  = (const float*)d_in[0];
  const float* k  = (const float*)d_in[1];
  const float* v  = (const float*)d_in[2];
  // d_in[3] = attn_mask: recomputed analytically on-device, not needed
  const float* Wq = (const float*)d_in[4];
  const float* bq = (const float*)d_in[5];
  const float* Wk = (const float*)d_in[6];
  const float* bk = (const float*)d_in[7];
  const float* Wv = (const float*)d_in[8];
  const float* bv = (const float*)d_in[9];
  const float* Wo = (const float*)d_in[10];
  const float* bo = (const float*)d_in[11];
  float* out      = (float*)d_out;
  _Float16* ws    = (_Float16*)d_ws;

  wtrans_kernel<<<dim3(1024, 4), dim3(256), 0, stream>>>(Wq, Wk, Wv, Wo, ws + OFF_WT);
  qkv_proj_kernel<<<dim3(32, 512, 3), dim3(32), 0, stream>>>(q, k, v, bq, bk, bv, ws);
  area_feat_kernel<<<dim3(32), dim3(256), 0, stream>>>(ws);
  area_attn_kernel<<<dim3(8, 128), dim3(128), 0, stream>>>(ws);
  out_proj_kernel<<<dim3(32, 512), dim3(32), 0, stream>>>(ws, bo, out);
}
